// GCLayer_22711787062030
// MI455X (gfx1250) — compile-verified, matrix operand order checked by reading, not run on hardware
//
#include <hip/hip_runtime.h>
#include <math.h>

// ---------- types ----------
typedef __attribute__((ext_vector_type(16))) __bf16        v16bf;
typedef __attribute__((ext_vector_type(16))) unsigned short v16u;
typedef __attribute__((ext_vector_type(8)))  float          v8f;

#define SROW 132   // LDS row stride in floats (128 padded -> avoids 64-bank conflicts)
#define DDIM 128

// ---------- helpers ----------
__device__ __forceinline__ unsigned short f2bf(float f) {
  unsigned u = __builtin_bit_cast(unsigned, f);
  u += 0x7FFFu + ((u >> 16) & 1u);           // round-to-nearest-even
  return (unsigned short)(u >> 16);
}

__device__ __forceinline__ float silu_f(float v) {
  return v / (1.f + __expf(-v));
}

// Hardware float atomic (CDNA5 GLOBAL_ATOMIC_ADD_F32, no-return -> STOREcnt).
// Guarantees the L2 atomic path instead of a CAS loop.
__device__ __forceinline__ void atomic_add_f32(float* p, float v) {
  asm volatile("global_atomic_add_f32 %0, %1, off" : : "v"(p), "v"(v) : "memory");
}

// Build 16x32 bf16 A-fragment (ISA 7.12.2 layout) from an LDS f32 tile.
// lanes 0-15: M=lane, halves 0..7 = K kt*32+0..7, halves 8..15 = K kt*32+16..23
// lanes 16-31: M=lane-16, K bases +8 / +24
__device__ __forceinline__ v16bf make_a_frag(const float* tile, int lane, int kt) {
  int m  = lane & 15;
  int b0 = kt * 32 + ((lane & 16) ? 8 : 0);
  const float* r = tile + m * SROW + b0;
  float4 f0 = *(const float4*)(r);
  float4 f1 = *(const float4*)(r + 4);
  float4 f2 = *(const float4*)(r + 16);
  float4 f3 = *(const float4*)(r + 20);
  v16u u;
  u[0]  = f2bf(f0.x); u[1]  = f2bf(f0.y); u[2]  = f2bf(f0.z); u[3]  = f2bf(f0.w);
  u[4]  = f2bf(f1.x); u[5]  = f2bf(f1.y); u[6]  = f2bf(f1.z); u[7]  = f2bf(f1.w);
  u[8]  = f2bf(f2.x); u[9]  = f2bf(f2.y); u[10] = f2bf(f2.z); u[11] = f2bf(f2.w);
  u[12] = f2bf(f3.x); u[13] = f2bf(f3.y); u[14] = f2bf(f3.z); u[15] = f2bf(f3.w);
  return __builtin_bit_cast(v16bf, u);
}

// B fragments are pre-swizzled by prep_weights into [kt][nt][lane][16] bf16.
__device__ __forceinline__ v16bf load_b_frag(const unsigned short* W, int kt, int nt, int lane) {
  return __builtin_bit_cast(v16bf,
      *(const v16u*)(W + (size_t)((((kt << 3) + nt) << 5) + lane) * 16));
}

// One wave computes a full 16x128 = (16 rows) x (8 column tiles) output block:
//   ct[nt] = A(16x128) @ W(128x128) tile nt  (+bias, optional SiLU)
template<bool SILU, bool HASBIAS>
__device__ __forceinline__ void gemm_tile(const float* tile, const unsigned short* W,
                                          const float* bias, int lane, v8f* ct) {
  v16bf a[4];
#pragma unroll
  for (int kt = 0; kt < 4; ++kt) a[kt] = make_a_frag(tile, lane, kt);
#pragma unroll
  for (int nt = 0; nt < 8; ++nt) {
    v8f c = {};
#pragma unroll
    for (int kt = 0; kt < 4; ++kt) {
      v16bf b = load_b_frag(W, kt, nt, lane);
      c = __builtin_amdgcn_wmma_f32_16x16x32_bf16(false, a[kt], false, b,
                                                  (short)0, c, false, false);
    }
    float bv = HASBIAS ? bias[nt * 16 + (lane & 15)] : 0.f;
#pragma unroll
    for (int r = 0; r < 8; ++r) {
      float v = c[r] + bv;
      if (SILU) v = silu_f(v);
      c[r] = v;
    }
    ct[nt] = c;
  }
}

// D-matrix layout (16x16 f32, 8 VGPRs): lane<16 -> rows 0..7, lane>=16 -> rows 8..15
__device__ __forceinline__ void store_ct_lds(float* buf, const v8f* ct, int lane) {
  int n0 = lane & 15, rb = (lane & 16) ? 8 : 0;
#pragma unroll
  for (int nt = 0; nt < 8; ++nt)
#pragma unroll
    for (int r = 0; r < 8; ++r)
      buf[(rb + r) * SROW + nt * 16 + n0] = ct[nt][r];
}

__device__ __forceinline__ void store_ct_global(float* g, int row0, const v8f* ct,
                                                int lane, int N) {
  int n0 = lane & 15, rb = (lane & 16) ? 8 : 0;
#pragma unroll
  for (int nt = 0; nt < 8; ++nt)
#pragma unroll
    for (int r = 0; r < 8; ++r) {
      int row = row0 + rb + r;
      if (row < N) g[(size_t)row * DDIM + nt * 16 + n0] = ct[nt][r];
    }
}

// LayerNorm over the wave's 16 LDS rows; lanes L and L+16 split row L
// into column halves and combine partials with one shuffle (full wave busy).
__device__ __forceinline__ void wave_ln(float* buf, const float* g, const float* b, int lane) {
  float* r  = buf + (lane & 15) * SROW;
  int    c0 = (lane & 16) ? 64 : 0;
  float  s = 0.f, s2 = 0.f;
#pragma unroll 4
  for (int c = c0; c < c0 + 64; ++c) { float v = r[c]; s += v; s2 += v * v; }
  s  += __shfl_xor(s, 16, 32);
  s2 += __shfl_xor(s2, 16, 32);
  float m  = s * (1.f / DDIM);
  float rs = rsqrtf(s2 * (1.f / DDIM) - m * m + 1e-5f);
#pragma unroll 4
  for (int c = c0; c < c0 + 64; ++c) r[c] = (r[c] - m) * rs * g[c] + b[c];
}

// ---------- kernel 0: convert weights to bf16 WMMA B-fragment layout ----------
// 7 matrices: lin_w, msg_w1, msg_w2, attW1[0:128], attW1[128:256], out_w1, out_w2
__global__ __launch_bounds__(1024) void prep_weights(
    const float* lin_w, const float* msg_w1, const float* msg_w2,
    const float* att_w1, const float* out_w1, const float* out_w2,
    unsigned short* dst) {
  const float* srcs[7] = { lin_w, msg_w1, msg_w2, att_w1, att_w1 + 128 * 128,
                           out_w1, out_w2 };
  const float* w = srcs[blockIdx.x];
  unsigned short* d = dst + (size_t)blockIdx.x * 16384;
  int t = threadIdx.x;
  int kt = t >> 8, nt = (t >> 5) & 7, lane = t & 31;
  int n  = nt * 16 + (lane & 15);
  int kb = kt * 32 + ((lane & 16) ? 8 : 0);
  unsigned short* o = d + (size_t)(((kt * 8 + nt) * 32 + lane)) * 16;
#pragma unroll
  for (int h = 0; h < 16; ++h) {
    int k = kb + ((h < 8) ? h : (h + 8));   // halves 8..15 -> K += 16
    o[h] = f2bf(w[k * 128 + n]);
  }
}

// ---------- kernel: zero msg accumulator ----------
__global__ void zero_kernel(float4* p, long n4) {
  long i = (long)blockIdx.x * blockDim.x + threadIdx.x;
  if (i < n4) p[i] = float4{0.f, 0.f, 0.f, 0.f};
}

// ---------- kernel 1: node pre-pass ----------
// x  = h @ lin_w + lin_b
// p  = x @ attW1_row + att_b1          (edge-att, row-half)
// q  = x @ attW1_col                   (edge-att, col-half)
// x_ = LN(silu(x @ msg_w1 + b1)) @ msg_w2 + b2
__global__ __launch_bounds__(128) void node_pre(
    const float* __restrict__ hin, const unsigned short* __restrict__ Wall,
    const float* lin_b, const float* msg_b1, const float* msg_ln_g,
    const float* msg_ln_b, const float* msg_b2, const float* att_b1,
    float* xo, float* x_o, float* po, float* qo, int N) {
  __shared__ __align__(16) float smem[4 * 16 * SROW];
  int wave = threadIdx.x >> 5, lane = threadIdx.x & 31;
  float* buf = smem + wave * 16 * SROW;
  int row0 = (blockIdx.x * 4 + wave) * 16;

  // stage 16x128 rows of h (coalesced float4 per lane)
#pragma unroll
  for (int r = 0; r < 16; ++r) {
    float4 v = {};
    if (row0 + r < N) v = ((const float4*)hin)[(size_t)(row0 + r) * 32 + lane];
    *(float4*)(buf + r * SROW + lane * 4) = v;
  }

  v8f ct[8];
  gemm_tile<false, true>(buf, Wall + 0 * 16384, lin_b, lane, ct);   // x
  store_ct_global(xo, row0, ct, lane, N);
  store_ct_lds(buf, ct, lane);                                      // buf := x

  gemm_tile<false, true>(buf, Wall + 3 * 16384, att_b1, lane, ct);  // p (b1 folded)
  store_ct_global(po, row0, ct, lane, N);

  gemm_tile<false, false>(buf, Wall + 4 * 16384, nullptr, lane, ct); // q
  store_ct_global(qo, row0, ct, lane, N);

  gemm_tile<true, true>(buf, Wall + 1 * 16384, msg_b1, lane, ct);   // t = silu(x@w1+b1)
  store_ct_lds(buf, ct, lane);                                      // buf := t
  wave_ln(buf, msg_ln_g, msg_ln_b, lane);

  gemm_tile<false, true>(buf, Wall + 2 * 16384, msg_b2, lane, ct);  // x_ = t@w2+b2
  store_ct_global(x_o, row0, ct, lane, N);
}

// ---------- kernel 2: edge pass (one wave per edge, L2-resident gathers) ----------
__global__ __launch_bounds__(256) void edge_kernel(
    const int* __restrict__ rowI, const int* __restrict__ colI,
    const float* __restrict__ ea, const float* __restrict__ em,
    const float* __restrict__ p, const float* __restrict__ q,
    const float* __restrict__ x_, const float* __restrict__ w1e,
    const float* __restrict__ w2, const float* __restrict__ b2p,
    float* __restrict__ msg, int E) {
  int e    = (int)((blockIdx.x * (unsigned)blockDim.x + threadIdx.x) >> 5);
  int lane = threadIdx.x & 31;
  if (e >= E) return;
  int   r = rowI[e], c = colI[e];
  float a = ea[e],   m = em[e];

  float4 pv  = ((const float4*)(p  + (size_t)r * DDIM))[lane];
  float4 qv  = ((const float4*)(q  + (size_t)c * DDIM))[lane];
  float4 wv  = ((const float4*)w1e)[lane];
  float4 w2v = ((const float4*)w2)[lane];

  float4 hv;
  hv.x = silu_f(pv.x + qv.x + a * wv.x);
  hv.y = silu_f(pv.y + qv.y + a * wv.y);
  hv.z = silu_f(pv.z + qv.z + a * wv.z);
  hv.w = silu_f(pv.w + qv.w + a * wv.w);

  float s = hv.x * w2v.x + hv.y * w2v.y + hv.z * w2v.z + hv.w * w2v.w;
#pragma unroll
  for (int o = 16; o > 0; o >>= 1) s += __shfl_xor(s, o, 32);

  float att = m / (1.f + __expf(-(s + b2p[0])));

  float4 xv = ((const float4*)(x_ + (size_t)c * DDIM))[lane];
  float* dst = msg + (size_t)r * DDIM + lane * 4;
  atomic_add_f32(dst + 0, xv.x * att);
  atomic_add_f32(dst + 1, xv.y * att);
  atomic_add_f32(dst + 2, xv.z * att);
  atomic_add_f32(dst + 3, xv.w * att);
}

// ---------- kernel 3: node post-pass ----------
// y = LN(silu(msg @ out_w1 + b1)) @ out_w2 + b2 ; out = LN(x + y)
__global__ __launch_bounds__(128) void node_post(
    const float* __restrict__ msg, const float* __restrict__ x,
    const unsigned short* __restrict__ Wall,
    const float* out_b1, const float* out_ln_g, const float* out_ln_b,
    const float* out_b2, const float* ln_g, const float* ln_b,
    float* __restrict__ out, int N) {
  __shared__ __align__(16) float smem[4 * 16 * SROW];
  int wave = threadIdx.x >> 5, lane = threadIdx.x & 31;
  float* buf = smem + wave * 16 * SROW;
  int row0 = (blockIdx.x * 4 + wave) * 16;

#pragma unroll
  for (int r = 0; r < 16; ++r) {
    float4 v = {};
    if (row0 + r < N) v = ((const float4*)msg)[(size_t)(row0 + r) * 32 + lane];
    *(float4*)(buf + r * SROW + lane * 4) = v;
  }

  v8f ct[8];
  gemm_tile<true, true>(buf, Wall + 5 * 16384, out_b1, lane, ct);   // silu(msg@w1+b1)
  store_ct_lds(buf, ct, lane);
  wave_ln(buf, out_ln_g, out_ln_b, lane);

  gemm_tile<false, true>(buf, Wall + 6 * 16384, out_b2, lane, ct);  // y
  store_ct_lds(buf, ct, lane);

  // residual + final LayerNorm: lanes L, L+16 split row L into column halves
  int row = row0 + (lane & 15);
  int c0  = (lane & 16) ? 64 : 0;
  float* br = buf + (lane & 15) * SROW;
  const float* xr = x + (size_t)row * DDIM;
  float s = 0.f, s2 = 0.f;
  if (row < N) {
#pragma unroll 4
    for (int c = c0; c < c0 + 64; ++c) {
      float v = br[c] + xr[c]; br[c] = v; s += v; s2 += v * v;
    }
  }
  s  += __shfl_xor(s, 16, 32);     // partner lane shares the same row/predicate
  s2 += __shfl_xor(s2, 16, 32);
  if (row < N) {
    float mn = s * (1.f / DDIM);
    float rs = rsqrtf(s2 * (1.f / DDIM) - mn * mn + 1e-5f);
    float* orow = out + (size_t)row * DDIM;
#pragma unroll 4
    for (int c = c0; c < c0 + 64; ++c)
      orow[c] = (br[c] - mn) * rs * ln_g[c] + ln_b[c];
  }
}

// ---------- host launcher ----------
extern "C" void kernel_launch(void* const* d_in, const int* in_sizes, int n_in,
                              void* d_out, int out_size, void* d_ws, size_t ws_size,
                              hipStream_t stream) {
  const float* h_in      = (const float*)d_in[0];
  const float* edge_attr = (const float*)d_in[1];
  const int*   rowI      = (const int*)d_in[2];
  const int*   colI      = (const int*)d_in[3];
  // d_in[4] node_mask: unused by reference
  const float* edge_mask = (const float*)d_in[5];
  const float* lin_w     = (const float*)d_in[6];
  const float* lin_b     = (const float*)d_in[7];
  const float* msg_w1    = (const float*)d_in[8];
  const float* msg_b1    = (const float*)d_in[9];
  const float* msg_ln_g  = (const float*)d_in[10];
  const float* msg_ln_b  = (const float*)d_in[11];
  const float* msg_w2    = (const float*)d_in[12];
  const float* msg_b2    = (const float*)d_in[13];
  const float* att_w1    = (const float*)d_in[14];   // [257,128]
  const float* att_b1    = (const float*)d_in[15];
  const float* att_w2    = (const float*)d_in[16];   // [128,1]
  const float* att_b2    = (const float*)d_in[17];   // [1]
  const float* out_w1    = (const float*)d_in[18];
  const float* out_b1    = (const float*)d_in[19];
  const float* out_ln_g  = (const float*)d_in[20];
  const float* out_ln_b  = (const float*)d_in[21];
  const float* out_w2    = (const float*)d_in[22];
  const float* out_b2    = (const float*)d_in[23];
  const float* ln_g      = (const float*)d_in[24];
  const float* ln_b      = (const float*)d_in[25];

  const int N = in_sizes[0] / DDIM;   // 50000
  const int E = in_sizes[2];          // 800000

  // workspace layout: [7x bf16 fragment matrices][x][x_][p][q][msg]
  unsigned short* Wall = (unsigned short*)d_ws;
  float* base = (float*)((char*)d_ws + 7 * 16384 * sizeof(unsigned short));
  size_t NP = (size_t)N * DDIM;
  float* x   = base;
  float* x_  = base + 1 * NP;
  float* p   = base + 2 * NP;
  float* q   = base + 3 * NP;
  float* msg = base + 4 * NP;

  prep_weights<<<7, 1024, 0, stream>>>(lin_w, msg_w1, msg_w2, att_w1,
                                       out_w1, out_w2, Wall);

  long n4 = (long)(NP / 4);
  zero_kernel<<<(unsigned)((n4 + 255) / 256), 256, 0, stream>>>((float4*)msg, n4);

  node_pre<<<(unsigned)((N + 63) / 64), 128, 0, stream>>>(
      h_in, Wall, lin_b, msg_b1, msg_ln_g, msg_ln_b, msg_b2, att_b1,
      x, x_, p, q, N);

  edge_kernel<<<(unsigned)((E + 7) / 8), 256, 0, stream>>>(
      rowI, colI, edge_attr, edge_mask, p, q, x_,
      att_w1 + 256 * 128, att_w2, att_b2, msg, E);

  node_post<<<(unsigned)((N + 63) / 64), 128, 0, stream>>>(
      msg, x, Wall, out_b1, out_ln_g, out_ln_b, out_b2, ln_g, ln_b,
      (float*)d_out, N);
}
